// GIN_VN_15350213116757
// MI455X (gfx1250) — matrix operand order repaired and checked by reading, not compiled, not run on hardware
//
#include <hip/hip_runtime.h>
#include <hip/hip_bf16.h>

// Problem constants (match reference)
#define NN_DIN 128
#define NN_H   256
#define NN_H4  1024
#define NN_DOUT 128
#define NN_G   512
#define NN_L   5

typedef __attribute__((ext_vector_type(16))) __bf16 v16bf;
typedef __attribute__((ext_vector_type(8)))  float  v8f;

union BFU { v16bf v; unsigned u[8]; };

__device__ __forceinline__ unsigned short f2bf(float x) {
  unsigned u = __float_as_uint(x);
  u = (u + 0x7FFFu + ((u >> 16) & 1u)) >> 16;   // round-to-nearest-even
  return (unsigned short)u;
}
__device__ __forceinline__ unsigned pack2bf(float a, float b) {
  return (unsigned)f2bf(a) | ((unsigned)f2bf(b) << 16);
}

// ---------------------------------------------------------------------------
// Weight transpose + fp32->bf16 convert:  W[K,N] fp32  ->  Wt[N,K] bf16
// ---------------------------------------------------------------------------
__global__ void wconv_kernel(const float* __restrict__ W,
                             unsigned short* __restrict__ Wt, int K, int N) {
  int t = blockIdx.x * blockDim.x + threadIdx.x;
  if (t >= K * N) return;
  int n = t / K;
  int k = t - n * K;
  Wt[t] = f2bf(W[(size_t)k * N + n]);
}

// ---------------------------------------------------------------------------
// Fused WMMA GEMM:  out = epilogue( act( (A [+A2|gather]) @ Wt^T + b1 [+b2] ) [+resid] )
//   A:   [M,K] fp32     Wt: [N,K] bf16 (pre-transposed)   out: [M,N] fp32
// Block tile 256x64, 8 waves (wave32), each wave 32(M)x64(N): 2 A-frags x 4 N-tiles.
// B tile staged via GLOBAL_LOAD_ASYNC_TO_LDS_B128 (CDNA5 async path, ASYNCcnt).
// LDS row stride 20 dwords: conflict-free (20r mod 64 distinct, r=0..15) and 16B aligned.
// ---------------------------------------------------------------------------
#define TM 256
#define TN 64
#define TK 32
#define LDP 20

__launch_bounds__(256)
__global__ void gemm_bf16_kernel(const float* __restrict__ A,
                                 const float* __restrict__ A2,
                                 const int*   __restrict__ rowidx,
                                 const unsigned* __restrict__ Wt,   // bf16 pairs
                                 const float* __restrict__ bias,
                                 const float* __restrict__ bias2,
                                 const float* __restrict__ resid,
                                 float* __restrict__ out,
                                 int M, int K, int Ncols, int gelu) {
  __shared__ unsigned Asl[TM * LDP];   // 20 KB
  __shared__ unsigned Bsl[TN * LDP];   // 5 KB

  const int tid   = threadIdx.x;
  const int lane  = tid & 31;
  const int wave  = tid >> 5;
  const int lrow  = lane & 15;
  const int lhalf = lane >> 4;
  const int m0 = blockIdx.x * TM;
  const int n0 = blockIdx.y * TN;
  const int Kd = K >> 1;               // Wt row length in dwords

  v8f zv = {0.f,0.f,0.f,0.f,0.f,0.f,0.f,0.f};
  v8f acc[2][4] = {{zv, zv, zv, zv}, {zv, zv, zv, zv}};

  // Async B-tile destination: raw LDS byte offset for this thread's 16B chunk
  const int brow  = tid >> 2;          // 0..63
  const int bpart = tid & 3;           // 4 dwords each
  unsigned bdst;
  {
    __attribute__((address_space(3))) unsigned* p3 =
        (__attribute__((address_space(3))) unsigned*)&Bsl[brow * LDP + bpart * 4];
    bdst = (unsigned)(unsigned long long)p3;
  }
  const unsigned long long gBrow =
      (unsigned long long)(const char*)(Wt + (size_t)(n0 + brow) * Kd + bpart * 4);

  for (int k0 = 0; k0 < K; k0 += TK) {
    // ---- async-copy B tile (bf16 weights) global -> LDS, tracked by ASYNCcnt
    {
      unsigned long long gB = gBrow + (unsigned long long)(k0 << 1); // (k0>>1) dwords
      asm volatile("global_load_async_to_lds_b128 %0, %1, off"
                   :: "v"(bdst), "v"(gB) : "memory");
    }
    // ---- stage A tile [256 x 32] fp32 -> bf16 LDS, one row per thread
    {
      int row  = tid;
      int grow = m0 + row;
      unsigned us[16];
      if (grow < M) {
        const float* ap = A + (size_t)grow * K + k0;
        const float* a2p = nullptr;
        if (A2) {
          int g2 = rowidx ? rowidx[grow] : grow;
          a2p = A2 + (size_t)g2 * K + k0;
        }
        #pragma unroll
        for (int c = 0; c < 8; ++c) {
          float4 f = *(const float4*)(ap + c * 4);
          if (A2) {
            float4 g = *(const float4*)(a2p + c * 4);
            f.x += g.x; f.y += g.y; f.z += g.z; f.w += g.w;
          }
          us[2*c]   = pack2bf(f.x, f.y);
          us[2*c+1] = pack2bf(f.z, f.w);
        }
      } else {
        #pragma unroll
        for (int c = 0; c < 16; ++c) us[c] = 0u;
      }
      #pragma unroll
      for (int c = 0; c < 4; ++c) {
        uint4 q; q.x = us[4*c]; q.y = us[4*c+1]; q.z = us[4*c+2]; q.w = us[4*c+3];
        *(uint4*)&Asl[row * LDP + c * 4] = q;   // 16B-aligned ds_store_b128
      }
    }
    asm volatile("s_wait_asynccnt 0x0" ::: "memory");
    __syncthreads();

    // ---- A fragments (16x32 bf16, ISA interleaved K map: dwords {0-3,8-11}+4*half)
    BFU a0, a1;
    {
      int ar0 = (wave * 32 + lrow) * LDP;
      int ar1 = ar0 + 16 * LDP;
      uint4 q;
      q = *(const uint4*)&Asl[ar0 + 4 * lhalf];
      a0.u[0]=q.x; a0.u[1]=q.y; a0.u[2]=q.z; a0.u[3]=q.w;
      q = *(const uint4*)&Asl[ar0 + 8 + 4 * lhalf];
      a0.u[4]=q.x; a0.u[5]=q.y; a0.u[6]=q.z; a0.u[7]=q.w;
      q = *(const uint4*)&Asl[ar1 + 4 * lhalf];
      a1.u[0]=q.x; a1.u[1]=q.y; a1.u[2]=q.z; a1.u[3]=q.w;
      q = *(const uint4*)&Asl[ar1 + 8 + 4 * lhalf];
      a1.u[4]=q.x; a1.u[5]=q.y; a1.u[6]=q.z; a1.u[7]=q.w;
    }
    // ---- 4 N-tiles: B fragment (32x16 bf16, ISA interleaved K map: dwords {0-3,8-11}+4*half) + 2 WMMAs each
    #pragma unroll
    for (int nt = 0; nt < 4; ++nt) {
      BFU bf;
      int br = (nt * 16 + lrow) * LDP + 4 * lhalf;
      uint4 q;
      q = *(const uint4*)&Bsl[br];
      bf.u[0]=q.x; bf.u[1]=q.y; bf.u[2]=q.z; bf.u[3]=q.w;
      q = *(const uint4*)&Bsl[br + 8];
      bf.u[4]=q.x; bf.u[5]=q.y; bf.u[6]=q.z; bf.u[7]=q.w;
      acc[0][nt] = __builtin_amdgcn_wmma_f32_16x16x32_bf16(
          false, a0.v, false, bf.v, (short)0, acc[0][nt], false, false);
      acc[1][nt] = __builtin_amdgcn_wmma_f32_16x16x32_bf16(
          false, a1.v, false, bf.v, (short)0, acc[1][nt], false, false);
    }
    __syncthreads();
  }

  // ---- epilogue: bias(+bias2) -> GELU -> +resid -> store
  #pragma unroll
  for (int mt = 0; mt < 2; ++mt) {
    #pragma unroll
    for (int nt = 0; nt < 4; ++nt) {
      int col = n0 + nt * 16 + lrow;
      float bv  = bias  ? bias[col]  : 0.f;
      float b2v = bias2 ? bias2[col] : 0.f;
      #pragma unroll
      for (int r = 0; r < 8; ++r) {
        int row = m0 + wave * 32 + mt * 16 + r + lhalf * 8;
        if (row < M) {
          float xv = acc[mt][nt][r] + bv + b2v;
          if (gelu) xv = 0.5f * xv * (1.0f + erff(xv * 0.70710678118654752f));
          if (resid) xv += resid[(size_t)row * Ncols + col];
          out[(size_t)row * Ncols + col] = xv;
        }
      }
    }
  }
}

// ---------------------------------------------------------------------------
// Edge scatter-add:  agg[dst[e], :] += h[src[e], :]   (64 threads/edge, 4 cols each)
// ---------------------------------------------------------------------------
__global__ void edge_scatter_kernel(const int* __restrict__ src,
                                    const int* __restrict__ dst,
                                    const float* __restrict__ h,
                                    float* __restrict__ agg, int E) {
  int t = blockIdx.x * blockDim.x + threadIdx.x;
  int e = t >> 6;
  if (e >= E) return;
  int c = (t & 63) * 4;
  int s = src[e], d = dst[e];
  float4 v = *(const float4*)(h + (size_t)s * NN_H + c);
  float* ap = agg + (size_t)d * NN_H + c;
  atomicAdd(ap + 0, v.x); atomicAdd(ap + 1, v.y);
  atomicAdd(ap + 2, v.z); atomicAdd(ap + 3, v.w);
}

// ---------------------------------------------------------------------------
// Segment mean pool: sum + count via atomics, then divide
// ---------------------------------------------------------------------------
__global__ void pool_sum_kernel(const float* __restrict__ h,
                                const int* __restrict__ batch,
                                float* __restrict__ psum,
                                float* __restrict__ cnt, int n) {
  int t = blockIdx.x * blockDim.x + threadIdx.x;
  int node = t >> 6;
  if (node >= n) return;
  int c = (t & 63) * 4;
  int b = batch[node];
  float4 v = *(const float4*)(h + (size_t)node * NN_H + c);
  float* pp = psum + (size_t)b * NN_H + c;
  atomicAdd(pp + 0, v.x); atomicAdd(pp + 1, v.y);
  atomicAdd(pp + 2, v.z); atomicAdd(pp + 3, v.w);
  if ((t & 63) == 0) atomicAdd(&cnt[b], 1.0f);
}

__global__ void pool_div_kernel(float* __restrict__ pool,
                                const float* __restrict__ cnt, int gh) {
  int t = blockIdx.x * blockDim.x + threadIdx.x;
  if (t >= gh) return;
  int g = t >> 8;   // H = 256
  pool[t] = pool[t] / fmaxf(cnt[g], 1.0f);
}

// ---------------------------------------------------------------------------
// BatchNorm over nodes (biased variance): stats (sum, sumsq per column) + apply
// ---------------------------------------------------------------------------
__global__ void bn_stats_kernel(const float* __restrict__ h,
                                float* __restrict__ stats, int n) {
  int c = threadIdx.x;                   // 256 columns
  int rows = (n + gridDim.x - 1) / gridDim.x;
  int r0 = blockIdx.x * rows;
  int r1 = r0 + rows; if (r1 > n) r1 = n;
  float s = 0.f, s2 = 0.f;
  for (int r = r0; r < r1; ++r) {
    float v = h[(size_t)r * NN_H + c];
    s += v; s2 += v * v;
  }
  atomicAdd(&stats[c], s);
  atomicAdd(&stats[NN_H + c], s2);
}

__global__ void bn_apply_kernel(float* __restrict__ h,
                                const float* __restrict__ stats,
                                const float* __restrict__ gamma,
                                const float* __restrict__ beta, int n) {
  int t = blockIdx.x * blockDim.x + threadIdx.x;
  if (t >= n * NN_H) return;
  int c = t & (NN_H - 1);
  float inv_n = 1.0f / (float)n;
  float mu  = stats[c] * inv_n;
  float var = stats[NN_H + c] * inv_n - mu * mu;
  h[t] = (h[t] - mu) * rsqrtf(var + 1e-5f) * gamma[c] + beta[c];
}

// ---------------------------------------------------------------------------
extern "C" void kernel_launch(void* const* d_in, const int* in_sizes, int n_in,
                              void* d_out, int out_size, void* d_ws, size_t ws_size,
                              hipStream_t stream) {
  (void)n_in; (void)out_size; (void)ws_size;
  // ---- inputs (setup_inputs order)
  const float* x       = (const float*)d_in[0];
  const int*   ei      = (const int*)  d_in[1];
  const int*   batch   = (const int*)  d_in[2];
  const float* pre_W1  = (const float*)d_in[3];
  const float* pre_b1  = (const float*)d_in[4];
  const float* pre_W2  = (const float*)d_in[5];
  const float* pre_b2  = (const float*)d_in[6];
  const float* gin_W1  = (const float*)d_in[7];
  const float* gin_b1  = (const float*)d_in[8];
  const float* gin_W2  = (const float*)d_in[9];
  const float* gin_b2  = (const float*)d_in[10];
  const float* ffn_W1  = (const float*)d_in[11];
  const float* ffn_b1  = (const float*)d_in[12];
  const float* ffn_W2  = (const float*)d_in[13];
  const float* ffn_b2  = (const float*)d_in[14];
  const float* upd_W1  = (const float*)d_in[15];
  const float* upd_b1  = (const float*)d_in[16];
  const float* upd_W2  = (const float*)d_in[17];
  const float* upd_b2  = (const float*)d_in[18];
  const float* prop_W1 = (const float*)d_in[19];
  const float* prop_b1 = (const float*)d_in[20];
  const float* prop_W2 = (const float*)d_in[21];
  const float* prop_b2 = (const float*)d_in[22];
  const float* bn_g    = (const float*)d_in[23];
  const float* bn_b    = (const float*)d_in[24];
  const float* vn      = (const float*)d_in[25];
  const float* post_W1 = (const float*)d_in[26];
  const float* post_b1 = (const float*)d_in[27];
  const float* post_W2 = (const float*)d_in[28];
  const float* post_b2 = (const float*)d_in[29];

  const int Nn = in_sizes[0] / NN_DIN;     // 50000
  const int E  = in_sizes[1] / 2;          // 800000
  const int* src = ei;
  const int* dst = ei + E;

  // ---- workspace carve
  size_t off = 0;
  auto carve = [&](size_t bytes) -> void* {
    off = (off + 255) & ~(size_t)255;
    void* p = (char*)d_ws + off;
    off += bytes;
    return p;
  };
  float* hA    = (float*)carve((size_t)Nn * NN_H  * 4);
  float* hB    = (float*)carve((size_t)Nn * NN_H  * 4);
  float* mid   = (float*)carve((size_t)Nn * NN_H  * 4);
  float* agg   = (float*)carve((size_t)Nn * NN_H  * 4);
  float* pool  = (float*)carve((size_t)NN_G * NN_H  * 4);
  float* cnt   = (float*)carve((size_t)NN_G * 4);
  float* midg  = (float*)carve((size_t)NN_G * NN_H  * 4);
  float* vbuf  = (float*)carve((size_t)NN_G * NN_H4 * 4);
  float* ovn   = (float*)carve((size_t)NN_G * NN_H  * 4);
  float* bnst  = (float*)carve((size_t)2 * NN_H * 4);

  const size_t bf_total =
      (size_t)NN_DIN * NN_H + (size_t)NN_H * NN_H +                      // pre
      (size_t)NN_L * (4 * (size_t)NN_H * NN_H +                         // gin+ffn
                      (size_t)NN_H * NN_H + (size_t)NN_H * NN_H4 +      // upd
                      (size_t)NN_H4 * NN_H + (size_t)NN_H * NN_H) +     // prop
      (size_t)NN_H * NN_H + (size_t)NN_H * NN_DOUT;                     // post
  unsigned short* wp = (unsigned short*)carve(bf_total * 2);

  auto conv = [&](const float* W, unsigned short* Wt, int K, int Ncols) {
    int tot = K * Ncols;
    wconv_kernel<<<(tot + 255) / 256, 256, 0, stream>>>(W, Wt, K, Ncols);
  };

  unsigned short* preW1t = wp; wp += NN_DIN * NN_H;  conv(pre_W1, preW1t, NN_DIN, NN_H);
  unsigned short* preW2t = wp; wp += NN_H * NN_H;    conv(pre_W2, preW2t, NN_H, NN_H);
  unsigned short *ginW1t[NN_L], *ginW2t[NN_L], *ffnW1t[NN_L], *ffnW2t[NN_L];
  unsigned short *updW1t[NN_L], *updW2t[NN_L], *propW1t[NN_L], *propW2t[NN_L];
  for (int i = 0; i < NN_L; ++i) {
    ginW1t[i]  = wp; wp += NN_H  * NN_H;  conv(gin_W1  + (size_t)i*NN_H*NN_H,   ginW1t[i],  NN_H,  NN_H);
    ginW2t[i]  = wp; wp += NN_H  * NN_H;  conv(gin_W2  + (size_t)i*NN_H*NN_H,   ginW2t[i],  NN_H,  NN_H);
    ffnW1t[i]  = wp; wp += NN_H  * NN_H;  conv(ffn_W1  + (size_t)i*NN_H*NN_H,   ffnW1t[i],  NN_H,  NN_H);
    ffnW2t[i]  = wp; wp += NN_H  * NN_H;  conv(ffn_W2  + (size_t)i*NN_H*NN_H,   ffnW2t[i],  NN_H,  NN_H);
    updW1t[i]  = wp; wp += NN_H  * NN_H;  conv(upd_W1  + (size_t)i*NN_H*NN_H,   updW1t[i],  NN_H,  NN_H);
    updW2t[i]  = wp; wp += NN_H  * NN_H4; conv(upd_W2  + (size_t)i*NN_H*NN_H4,  updW2t[i],  NN_H,  NN_H4);
    propW1t[i] = wp; wp += NN_H4 * NN_H;  conv(prop_W1 + (size_t)i*NN_H4*NN_H,  propW1t[i], NN_H4, NN_H);
    propW2t[i] = wp; wp += NN_H  * NN_H;  conv(prop_W2 + (size_t)i*NN_H*NN_H,   propW2t[i], NN_H,  NN_H);
  }
  unsigned short* postW1t = wp; wp += NN_H * NN_H;    conv(post_W1, postW1t, NN_H, NN_H);
  unsigned short* postW2t = wp; wp += NN_H * NN_DOUT; conv(post_W2, postW2t, NN_H, NN_DOUT);

  auto gemm = [&](const float* A, const float* A2, const int* ridx,
                  const unsigned short* Wt, const float* b1, const float* b2,
                  const float* resid, float* outp, int M, int K, int Ncols, int gelu) {
    dim3 grid((M + TM - 1) / TM, Ncols / TN);
    gemm_bf16_kernel<<<grid, 256, 0, stream>>>(A, A2, ridx, (const unsigned*)Wt,
                                               b1, b2, resid, outp, M, K, Ncols, gelu);
  };

  // ---- pre-FFNN: h = GELU(x@W1+b1)@W2+b2
  gemm(x,   nullptr, nullptr, preW1t, pre_b1, nullptr, nullptr, mid, Nn, NN_DIN, NN_H, 1);
  gemm(mid, nullptr, nullptr, preW2t, pre_b2, nullptr, nullptr, hA,  Nn, NN_H,   NN_H, 0);

  for (int i = 0; i < NN_L; ++i) {
    // mean pool over graphs
    hipMemsetAsync(pool, 0, (size_t)NN_G * NN_H * 4, stream);
    hipMemsetAsync(cnt,  0, (size_t)NN_G * 4, stream);
    pool_sum_kernel<<<((size_t)Nn * 64 + 255) / 256, 256, 0, stream>>>(hA, batch, pool, cnt, Nn);
    pool_div_kernel<<<(NN_G * NN_H + 255) / 256, 256, 0, stream>>>(pool, cnt, NN_G * NN_H);

    // v = vn + FFNN_upd(pool);  outvn = FFNN_prop(v)
    gemm(pool, nullptr, nullptr, updW1t[i],  upd_b1 + (size_t)i*NN_H,  nullptr, nullptr, midg, NN_G, NN_H,  NN_H,  1);
    gemm(midg, nullptr, nullptr, updW2t[i],  upd_b2 + (size_t)i*NN_H4, vn,      nullptr, vbuf, NN_G, NN_H,  NN_H4, 0);
    gemm(vbuf, nullptr, nullptr, propW1t[i], prop_b1 + (size_t)i*NN_H, nullptr, nullptr, midg, NN_G, NN_H4, NN_H,  1);
    gemm(midg, nullptr, nullptr, propW2t[i], prop_b2 + (size_t)i*NN_H, nullptr, nullptr, ovn,  NN_G, NN_H,  NN_H,  0);

    // edge aggregation: agg[d] += h[s]
    hipMemsetAsync(agg, 0, (size_t)Nn * NN_H * 4, stream);
    edge_scatter_kernel<<<((size_t)E * 64 + 255) / 256, 256, 0, stream>>>(src, dst, hA, agg, E);

    // GINConv + residual: hB = FFNN_gin(hA + agg) + hA
    gemm(hA,  agg,     nullptr, ginW1t[i], gin_b1 + (size_t)i*NN_H, nullptr, nullptr, mid, Nn, NN_H, NN_H, 1);
    gemm(mid, nullptr, nullptr, ginW2t[i], gin_b2 + (size_t)i*NN_H, nullptr, hA,      hB,  Nn, NN_H, NN_H, 0);

    // BatchNorm (batch statistics, biased var), in place on hB
    hipMemsetAsync(bnst, 0, (size_t)2 * NN_H * 4, stream);
    bn_stats_kernel<<<512, NN_H, 0, stream>>>(hB, bnst, Nn);
    bn_apply_kernel<<<((size_t)Nn * NN_H + 255) / 256, 256, 0, stream>>>(
        hB, bnst, bn_g + (size_t)i*NN_H, bn_b + (size_t)i*NN_H, Nn);

    // hA = FFNN_ffn(outvn[batch] + hB) + hB
    gemm(hB,  ovn,     batch,   ffnW1t[i], ffn_b1 + (size_t)i*NN_H, nullptr, nullptr, mid, Nn, NN_H, NN_H, 1);
    gemm(mid, nullptr, nullptr, ffnW2t[i], ffn_b2 + (size_t)i*NN_H, nullptr, hB,      hA,  Nn, NN_H, NN_H, 0);
  }

  // ---- post: out = FFNN_post(mean_pool(h))
  hipMemsetAsync(pool, 0, (size_t)NN_G * NN_H * 4, stream);
  hipMemsetAsync(cnt,  0, (size_t)NN_G * 4, stream);
  pool_sum_kernel<<<((size_t)Nn * 64 + 255) / 256, 256, 0, stream>>>(hA, batch, pool, cnt, Nn);
  pool_div_kernel<<<(NN_G * NN_H + 255) / 256, 256, 0, stream>>>(pool, cnt, NN_G * NN_H);
  gemm(pool, nullptr, nullptr, postW1t, post_b1, nullptr, nullptr, midg,          NN_G, NN_H, NN_H,    1);
  gemm(midg, nullptr, nullptr, postW2t, post_b2, nullptr, nullptr, (float*)d_out, NN_G, NN_H, NN_DOUT, 0);
}
